// GatedMemoryModule_80590766342421
// MI455X (gfx1250) — matrix operand order repaired and checked by reading, not compiled, not run on hardware
//
#include <hip/hip_runtime.h>
#include <hip/hip_bf16.h>

typedef __attribute__((ext_vector_type(16))) _Float16 v16h;
typedef __attribute__((ext_vector_type(8)))  _Float16 v8h;
typedef __attribute__((ext_vector_type(8)))  float    v8f;
typedef __attribute__((ext_vector_type(4)))  float    v4f;

#define B_    8
#define T_    2048
#define D_    256
#define N_    2048
#define ROWS  32
// (1/temperature) * log2(e) = 10 * 1.4426950408889634
#define INV_T_LOG2E 14.426950408889634f

union H16 { v16h v; v8h h[2]; };

// ---------------------------------------------------------------------------
// Prep: build f16 B-operand-ready packs of the memory matrix.
//   b1: for GEMM1  (B = memory^T chunks, 32x16 KxN tiles)
//       layout [ntile(128)][kc(8)][lane(32)][e(16)]
//       b1[...] = memory[ntile*16 + (lane&15)][kc*32 + (lane>>4)*16 + e]
//   b2: for GEMM2  (B = memory chunks)
//       layout [dtile(16)][kc(64)][lane(32)][e(16)]
//       b2[...] = memory[kc*32 + (lane>>4)*16 + e][dtile*16 + (lane&15)]
// Each is 524288 halves (1 MB); both stay resident in L2.
// ---------------------------------------------------------------------------
__global__ __launch_bounds__(256) void gm_pack(const float* __restrict__ mem,
                                               _Float16* __restrict__ b1,
                                               _Float16* __restrict__ b2) {
  int idx  = blockIdx.x * 256 + threadIdx.x;       // 0 .. 524287
  int e    = idx & 15;
  int lane = (idx >> 4) & 31;
  {
    int kc = (idx >> 9) & 7;
    int nt = idx >> 12;                            // 0..127
    int n  = nt * 16 + (lane & 15);
    int k  = kc * 32 + ((lane >> 4) << 4) + e;
    b1[idx] = (_Float16)mem[(size_t)n * D_ + k];
  }
  {
    int kc = (idx >> 9) & 63;
    int dt = idx >> 15;                            // 0..15
    int d  = dt * 16 + (lane & 15);
    int k  = kc * 32 + ((lane >> 4) << 4) + e;
    b2[idx] = (_Float16)mem[(size_t)k * D_ + d];
  }
}

// ---------------------------------------------------------------------------
// Fused: scores GEMM -> softmax -> attn_w out -> retrieve GEMM -> concat out.
// One block = 32 query rows, 8 waves (wave32). Each wave owns one M-tile and
// a contiguous strip of N-tiles, processed 4 tiles per A-load with 4
// independent WMMA accumulator chains.
// Dynamic LDS: S[32*2048] f32 | Q[32*256] f16 | red[256] | rmax[32] | rinv[32]
// ---------------------------------------------------------------------------
__global__ __launch_bounds__(256) void gm_fused(const float* __restrict__ q,
                                                const _Float16* __restrict__ b1,
                                                const _Float16* __restrict__ b2,
                                                float* __restrict__ outq,
                                                float* __restrict__ outw) {
  extern __shared__ char smem[];
  float*    S    = (float*)smem;                                  // 262144 B
  _Float16* Q    = (_Float16*)(smem + ROWS * N_ * 4);             //  16384 B
  float*    red  = (float*)(smem + ROWS * N_ * 4 + ROWS * D_ * 2);
  float*    rmax = red + 256;
  float*    rinv = rmax + ROWS;

  const int tid     = threadIdx.x;
  const int lane    = tid & 31;
  const int wave    = tid >> 5;
  const int rowBase = blockIdx.x * ROWS;

  // --- stage queries (f32 -> f16 in LDS) + pass-through first half of outq
  #pragma unroll
  for (int i = 0; i < (ROWS * D_) / 256; ++i) {
    int gi = i * 256 + tid;
    int r = gi >> 8, c = gi & (D_ - 1);
    float val = q[(size_t)(rowBase + r) * D_ + c];
    Q[gi] = (_Float16)val;
    outq[(size_t)(rowBase + r) * (2 * D_) + c] = val;
  }
  __syncthreads();

  const int mt   = wave & 1;          // M-tile (0..1)
  const int wgrp = wave >> 1;         // 0..3
  const int ml   = lane & 15;
  const int koff = (lane >> 4) * 16;  // A-operand K split across lane halves
  const int rhalf = (lane >> 4) * 8;  // C/D row split across lane halves

  // --- GEMM1: S[32 x 2048] = Q * memory^T  (f16 WMMA, f32 acc)
  //     wave-group owns nt in [wgrp*32, wgrp*32+32); 4 N-tiles per A-load.
  for (int j = 0; j < 8; ++j) {
    int nt0 = (wgrp << 5) + (j << 2);
    v8f acc0 = {}, acc1 = {}, acc2 = {}, acc3 = {};
    const _Float16* ap = Q + (mt * 16 + ml) * D_ + koff;
    const _Float16* bp = b1 + (size_t)nt0 * 4096 + lane * 16;
    #pragma unroll
    for (int kc = 0; kc < 8; ++kc) {
      H16 a;
      a.h[0] = *(const v8h*)(ap + kc * 32);
      a.h[1] = *(const v8h*)(ap + kc * 32 + 8);
      H16 b;
      b.h[0] = *(const v8h*)(bp + kc * 512);
      b.h[1] = *(const v8h*)(bp + kc * 512 + 8);
      acc0 = __builtin_amdgcn_wmma_f32_16x16x32_f16(false, a.v, false, b.v,
                                                    (short)0, acc0, false, false);
      b.h[0] = *(const v8h*)(bp + 4096 + kc * 512);
      b.h[1] = *(const v8h*)(bp + 4096 + kc * 512 + 8);
      acc1 = __builtin_amdgcn_wmma_f32_16x16x32_f16(false, a.v, false, b.v,
                                                    (short)0, acc1, false, false);
      b.h[0] = *(const v8h*)(bp + 8192 + kc * 512);
      b.h[1] = *(const v8h*)(bp + 8192 + kc * 512 + 8);
      acc2 = __builtin_amdgcn_wmma_f32_16x16x32_f16(false, a.v, false, b.v,
                                                    (short)0, acc2, false, false);
      b.h[0] = *(const v8h*)(bp + 12288 + kc * 512);
      b.h[1] = *(const v8h*)(bp + 12288 + kc * 512 + 8);
      acc3 = __builtin_amdgcn_wmma_f32_16x16x32_f16(false, a.v, false, b.v,
                                                    (short)0, acc3, false, false);
    }
    int r0 = mt * 16 + rhalf;
    #pragma unroll
    for (int v = 0; v < 8; ++v) {
      float* srow = S + (r0 + v) * N_ + nt0 * 16 + ml;
      srow[0]  = acc0[v];
      srow[16] = acc1[v];
      srow[32] = acc2[v];
      srow[48] = acc3[v];
    }
  }
  __syncthreads();

  // --- softmax over N (8 threads per row); 1/T folded into exp2 scale
  {
    int row = tid >> 3, sub = tid & 7;
    v4f* sr = (v4f*)(S + row * N_ + sub * 256);
    float m = -3.0e38f;
    #pragma unroll 4
    for (int j = 0; j < 64; ++j) {
      v4f x = sr[j];
      m = fmaxf(m, fmaxf(fmaxf(x[0], x[1]), fmaxf(x[2], x[3])));
    }
    red[tid] = m;
    __syncthreads();
    if (sub == 0) {
      float mm = red[tid];
      #pragma unroll
      for (int k = 1; k < 8; ++k) mm = fmaxf(mm, red[tid + k]);
      rmax[row] = mm;
    }
    __syncthreads();
    float mm  = rmax[row];
    float sum = 0.f;
    #pragma unroll 4
    for (int j = 0; j < 64; ++j) {
      v4f x = sr[j];
      v4f e;
      e[0] = exp2f((x[0] - mm) * INV_T_LOG2E);
      e[1] = exp2f((x[1] - mm) * INV_T_LOG2E);
      e[2] = exp2f((x[2] - mm) * INV_T_LOG2E);
      e[3] = exp2f((x[3] - mm) * INV_T_LOG2E);
      sum += (e[0] + e[1]) + (e[2] + e[3]);
      sr[j] = e;
    }
    red[tid] = sum;
    __syncthreads();
    if (sub == 0) {
      float s = red[tid];
      #pragma unroll
      for (int k = 1; k < 8; ++k) s += red[tid + k];
      rinv[row] = 1.0f / s;
    }
    __syncthreads();
  }

  // --- write attn_w (normalized), coalesced float4, non-temporal (134 MB
  //     stream that is never re-read; keep L2 for the B packs)
  #pragma unroll 2
  for (int c = 0; c < 64; ++c) {
    int gi  = c * 256 + tid;            // float4 index, 16384 total
    int row = gi >> 9, c4 = gi & 511;
    v4f x = ((const v4f*)S)[gi];
    x *= rinv[row];
    __builtin_nontemporal_store(x, ((v4f*)(outw + (size_t)(rowBase + row) * N_)) + c4);
  }

  // --- GEMM2: retrieved[32 x 256] = softmax(S) * memory.
  //     Row-wise 1/sum is applied in the epilogue (distributes over the dot),
  //     so the K-loop A-conversion is a pure f32->f16 cvt shared by all 4
  //     D-tiles (4 independent accumulator chains).
  {
    v8f acc0 = {}, acc1 = {}, acc2 = {}, acc3 = {};
    const float*    srow = S + (mt * 16 + ml) * N_ + koff;
    const _Float16* bp   = b2 + (size_t)wgrp * 32768 + lane * 16;  // dt = wgrp + 4t
    for (int kc = 0; kc < 64; ++kc) {
      const float* sp = srow + kc * 32;
      v4f s0 = *(const v4f*)sp;
      v4f s1 = *(const v4f*)(sp + 4);
      v4f s2 = *(const v4f*)(sp + 8);
      v4f s3 = *(const v4f*)(sp + 12);
      H16 a;
      #pragma unroll
      for (int e2 = 0; e2 < 4; ++e2) {
        a.v[e2]      = (_Float16)s0[e2];
        a.v[4 + e2]  = (_Float16)s1[e2];
        a.v[8 + e2]  = (_Float16)s2[e2];
        a.v[12 + e2] = (_Float16)s3[e2];
      }
      const _Float16* bk = bp + kc * 512;
      H16 b;
      b.h[0] = *(const v8h*)(bk);
      b.h[1] = *(const v8h*)(bk + 8);
      acc0 = __builtin_amdgcn_wmma_f32_16x16x32_f16(false, a.v, false, b.v,
                                                    (short)0, acc0, false, false);
      b.h[0] = *(const v8h*)(bk + 131072);
      b.h[1] = *(const v8h*)(bk + 131072 + 8);
      acc1 = __builtin_amdgcn_wmma_f32_16x16x32_f16(false, a.v, false, b.v,
                                                    (short)0, acc1, false, false);
      b.h[0] = *(const v8h*)(bk + 262144);
      b.h[1] = *(const v8h*)(bk + 262144 + 8);
      acc2 = __builtin_amdgcn_wmma_f32_16x16x32_f16(false, a.v, false, b.v,
                                                    (short)0, acc2, false, false);
      b.h[0] = *(const v8h*)(bk + 393216);
      b.h[1] = *(const v8h*)(bk + 393216 + 8);
      acc3 = __builtin_amdgcn_wmma_f32_16x16x32_f16(false, a.v, false, b.v,
                                                    (short)0, acc3, false, false);
    }
    int r0 = mt * 16 + rhalf;
    float rv[8];
    #pragma unroll
    for (int v = 0; v < 8; ++v) rv[v] = rinv[r0 + v];
    #pragma unroll
    for (int v = 0; v < 8; ++v) {
      float* orow = outq + (size_t)(rowBase + r0 + v) * (2 * D_) + D_ + ml;
      orow[(wgrp + 0)  * 16] = acc0[v] * rv[v];
      orow[(wgrp + 4)  * 16] = acc1[v] * rv[v];
      orow[(wgrp + 8)  * 16] = acc2[v] * rv[v];
      orow[(wgrp + 12) * 16] = acc3[v] * rv[v];
    }
  }
}

// ---------------------------------------------------------------------------
extern "C" void kernel_launch(void* const* d_in, const int* in_sizes, int n_in,
                              void* d_out, int out_size, void* d_ws, size_t ws_size,
                              hipStream_t stream) {
  const float* queries = (const float*)d_in[0];   // [8, 2048, 256]
  const float* memory  = (const float*)d_in[1];   // [2048, 256]
  float* outq = (float*)d_out;                            // [8,2048,512]
  float* outw = (float*)d_out + (size_t)B_ * T_ * 2 * D_; // [8,2048,2048]

  _Float16* b1 = (_Float16*)d_ws;                 // 1 MB
  _Float16* b2 = b1 + (size_t)N_ * D_;            // 1 MB

  gm_pack<<<(N_ * D_) / 256, 256, 0, stream>>>(memory, b1, b2);

  const int shmem = ROWS * N_ * 4 + ROWS * D_ * 2 + (256 + ROWS + ROWS) * 4;
  (void)hipFuncSetAttribute((const void*)gm_fused,
                            hipFuncAttributeMaxDynamicSharedMemorySize, shmem);
  gm_fused<<<(B_ * T_) / ROWS, 256, shmem, stream>>>(queries, b1, b2, outq, outw);
}
